// CNN_FS_D2ST_SRC_29283087024824
// MI455X (gfx1250) — matrix operand order repaired and checked by reading, not compiled, not run on hardware
//
#include <hip/hip_runtime.h>
#include <hip/hip_bf16.h>

typedef __attribute__((ext_vector_type(16))) __bf16 v16bf;
typedef __attribute__((ext_vector_type(8)))  float  v8f;

#define NQ        8192
#define NSUP      640
#define NCLS      64
#define TT        16
#define DD        256
#define ROWLEN    (TT * DD)      // 4096 elements per sample/class

// ---------------- workspace layout (bytes) ----------------
#define WS_SUMS   0              // proto sums f32 [64][16][256]   (1 MB)
#define WS_CNT    1048576        // counts f32 [64]
#define WS_PN     1048832        // pn bf16 [64][16][256]          (512 KB)
#define WS_QINV   1573120        // query inv-norms f32 [8192*16]  (512 KB)

// -------- kernel 1: segment-sum prototypes via atomics --------
__global__ void proto_accum_k(const float* __restrict__ sup,
                              const int* __restrict__ labels,
                              float* __restrict__ sums,
                              float* __restrict__ counts) {
    long idx = (long)blockIdx.x * 256 + threadIdx.x;   // 640*4096 total
    int s = (int)(idx >> 12);
    int e = (int)(idx & 4095);
    if (s >= NSUP) return;
    int lab = labels[s];
    atomicAdd(&sums[(long)lab * ROWLEN + e], sup[idx]);
    if (e == 0) atomicAdd(&counts[lab], 1.0f);
}

// -------- block (256-thread) sum reduce helper --------
__device__ __forceinline__ float block_sum_256(float v) {
    #pragma unroll
    for (int m = 16; m >= 1; m >>= 1) v += __shfl_xor(v, m, 32);
    __shared__ float parts[8];
    int w = threadIdx.x >> 5, lane = threadIdx.x & 31;
    if (lane == 0) parts[w] = v;
    __syncthreads();
    float total = 0.f;
    #pragma unroll
    for (int i = 0; i < 8; ++i) total += parts[i];
    return total;
}

// -------- kernel 2: mean + L2-normalize protos -> bf16 --------
__global__ void proto_norm_k(const float* __restrict__ sums,
                             const float* __restrict__ counts,
                             __bf16* __restrict__ pn) {
    int row = blockIdx.x;                 // c*16 + s, 1024 rows
    int c = row >> 4;
    float cnt = fmaxf(counts[c], 1.0f);
    float v = sums[(long)row * DD + threadIdx.x] / cnt;
    float ss = block_sum_256(v * v);
    float inv = 1.0f / fmaxf(sqrtf(ss), 1e-12f);
    pn[(long)row * DD + threadIdx.x] = (__bf16)(v * inv);
}

// -------- kernel 3: query row inverse L2 norms --------
__global__ void query_norm_k(const float* __restrict__ qf,
                             float* __restrict__ qinv) {
    int row = blockIdx.x;                 // q*16 + t, 131072 rows
    float v = qf[(long)row * DD + threadIdx.x];
    float ss = block_sum_256(v * v);
    if (threadIdx.x == 0) qinv[row] = 1.0f / fmaxf(sqrtf(ss), 1e-12f);
}

// -------- VALU DPP butterflies within 16-lane rows (co-execute with XDL) ----
template <int CTRL>
__device__ __forceinline__ float fmax_dpp(float x) {
    int t = __builtin_amdgcn_update_dpp(0, __float_as_int(x), CTRL, 0xF, 0xF, true);
    return fmaxf(x, __int_as_float(t));
}
template <int CTRL>
__device__ __forceinline__ float fadd_dpp(float x) {
    int t = __builtin_amdgcn_update_dpp(0, __float_as_int(x), CTRL, 0xF, 0xF, true);
    return x + __int_as_float(t);
}
__device__ __forceinline__ float row_max16(float x) {
    x = fmax_dpp<0xB1>(x);    // quad_perm(1,0,3,2)  : xor 1
    x = fmax_dpp<0x4E>(x);    // quad_perm(2,3,0,1)  : xor 2
    x = fmax_dpp<0x141>(x);   // ROW_HALF_MIRROR     : 8-lane halves
    x = fmax_dpp<0x140>(x);   // ROW_MIRROR          : full 16-lane row
    return x;
}
__device__ __forceinline__ float row_sum16(float x) {
    x = fadd_dpp<0xB1>(x);
    x = fadd_dpp<0x4E>(x);
    x = fadd_dpp<0x141>(x);
    x = fadd_dpp<0x140>(x);
    return x;
}

// -------- kernel 4: WMMA bf16 similarity + bi-MHM reduction --------
// One wave per query; 8 waves/block share double-buffered per-class LDS tiles.
__global__ void __launch_bounds__(256)
bimhm_wmma_k(const float* __restrict__ qf,
             const float* __restrict__ qinv,
             const __bf16* __restrict__ pn,
             float* __restrict__ out) {
    __shared__ __align__(32) __bf16 ldsB[2][TT * DD];  // 2 x 8 KB class tiles

    const int lane = threadIdx.x & 31;
    const int half = lane >> 4;          // 0: lanes 0-15, 1: lanes 16-31
    const int tq   = lane & 15;          // matrix row (Tq) / col (Ts) index
    const int q    = blockIdx.x * 8 + (threadIdx.x >> 5);

    // ---- preload + normalize + pack A fragments (16x256 bf16, 64 VGPRs) ----
    const float* qrow = qf + (long)q * ROWLEN + (long)tq * DD;
    const float  inv  = qinv[q * TT + tq];
    v16bf afr[8];
    #pragma unroll
    for (int kk = 0; kk < 8; ++kk) {
        const int k0 = kk * 32;
        // lanes 0-15 hold K = k0+{0..15}; lanes 16-31: k0+{16..31}
        const float4* p0 = (const float4*)(qrow + k0 + half * 16);
        const float4* p1 = (const float4*)(qrow + k0 + half * 16 + 8);
        float4 x0 = p0[0], x1 = p0[1];
        float4 y0 = p1[0], y1 = p1[1];
        v16bf a;
        a[0]  = (__bf16)(x0.x * inv); a[1]  = (__bf16)(x0.y * inv);
        a[2]  = (__bf16)(x0.z * inv); a[3]  = (__bf16)(x0.w * inv);
        a[4]  = (__bf16)(x1.x * inv); a[5]  = (__bf16)(x1.y * inv);
        a[6]  = (__bf16)(x1.z * inv); a[7]  = (__bf16)(x1.w * inv);
        a[8]  = (__bf16)(y0.x * inv); a[9]  = (__bf16)(y0.y * inv);
        a[10] = (__bf16)(y0.z * inv); a[11] = (__bf16)(y0.w * inv);
        a[12] = (__bf16)(y1.x * inv); a[13] = (__bf16)(y1.y * inv);
        a[14] = (__bf16)(y1.z * inv); a[15] = (__bf16)(y1.w * inv);
        afr[kk] = a;
    }

    // ---- prologue: stage class tile 0 into LDS buffer 0 ----
    {
        const uint4* src = (const uint4*)pn;
        uint4* dst = (uint4*)ldsB[0];
        dst[threadIdx.x]       = src[threadIdx.x];
        dst[threadIdx.x + 256] = src[threadIdx.x + 256];
    }
    __syncthreads();

    for (int c = 0; c < NCLS; ++c) {
        const bool more = (c + 1 < NCLS);

        // issue next class tile's global loads (latency hides under WMMA)
        uint4 stg0, stg1;
        if (more) {
            const uint4* src = (const uint4*)(pn + (long)(c + 1) * ROWLEN);
            stg0 = src[threadIdx.x];
            stg1 = src[threadIdx.x + 256];
        }
        // prefetch class c+2 toward L2
        if (c + 2 < NCLS && threadIdx.x < 64)
            __builtin_prefetch((const char*)(pn + (long)(c + 2) * ROWLEN)
                               + threadIdx.x * 128, 0, 1);

        // ---- preload all 8 B fragments, then 8 back-to-back WMMAs ----
        const __bf16* buf = ldsB[c & 1];
        v16bf bfr[8];
        #pragma unroll
        for (int kk = 0; kk < 8; ++kk)
            bfr[kk] = *(const v16bf*)(&buf[tq * DD + kk * 32 + half * 16]);

        v8f acc = {};
        #pragma unroll
        for (int kk = 0; kk < 8; ++kk)
            acc = __builtin_amdgcn_wmma_f32_16x16x32_bf16(
                      false, afr[kk], false, bfr[kk], (short)0, acc, false, false);

        // ---- bi-MHM reduction (VALU DPP; only xor-16 uses the DS pipe) ----
        // C layout: VGPR r, lanes 0-15 -> (M=r, N=lane); lanes 16-31 -> (M=r+8)
        float rs = 0.f;
        #pragma unroll
        for (int r = 0; r < 8; ++r) rs += row_max16(acc[r]);   // sum_t max_s
        rs += __shfl_xor(rs, 16, 32);                          // both halves

        float cm = acc[0];
        #pragma unroll
        for (int r = 1; r < 8; ++r) cm = fmaxf(cm, acc[r]);    // max over M half
        cm = fmaxf(cm, __shfl_xor(cm, 16, 32));                // max_t per N
        cm = row_sum16(cm);                                    // sum over 16 s

        if (lane == 0)
            out[(long)q * NCLS + c] = 32.0f - rs - cm;

        // write next tile into the other LDS buffer, one barrier per class
        if (more) {
            uint4* dst = (uint4*)ldsB[(c + 1) & 1];
            dst[threadIdx.x]       = stg0;
            dst[threadIdx.x + 256] = stg1;
        }
        __syncthreads();
    }
}

extern "C" void kernel_launch(void* const* d_in, const int* in_sizes, int n_in,
                              void* d_out, int out_size, void* d_ws, size_t ws_size,
                              hipStream_t stream) {
    const float* sup    = (const float*)d_in[0];
    const float* qf     = (const float*)d_in[1];
    const int*   labels = (const int*)d_in[2];
    float* out = (float*)d_out;

    char* ws = (char*)d_ws;
    float*  proto_sums = (float*)(ws + WS_SUMS);
    float*  counts     = (float*)(ws + WS_CNT);
    __bf16* pn         = (__bf16*)(ws + WS_PN);
    float*  qinv       = (float*)(ws + WS_QINV);

    (void)hipMemsetAsync(ws, 0, WS_PN, stream);  // zero sums + counts (graph-safe)

    proto_accum_k<<<(NSUP * ROWLEN) / 256, 256, 0, stream>>>(sup, labels,
                                                             proto_sums, counts);
    proto_norm_k<<<NCLS * TT, 256, 0, stream>>>(proto_sums, counts, pn);
    query_norm_k<<<NQ * TT, 256, 0, stream>>>(qf, qinv);
    bimhm_wmma_k<<<NQ / 8, 256, 0, stream>>>(qf, qinv, pn, out);
}
